// NewRN_18528488915056
// MI455X (gfx1250) — matrix operand order repaired and checked by reading, not compiled, not run on hardware
//
#include <hip/hip_runtime.h>
#include <hip/hip_bf16.h>

// ---------------------------------------------------------------------------
// Relation-Network few-shot pipeline for MI455X (gfx1250, wave32, WMMA + TDM).
//
// Dominant cost: 307200 rows x (256x256 GEMM -> relu) x 3, fused per-tile in
// LDS using v_wmma_f32_16x16x32_bf16. Weights are staged L2 -> LDS by the
// Tensor Data Mover (tensor_load_to_lds) with descriptor-driven padding that
// reproduces the 264-element bank-conflict-free LDS row stride.
// Conv stack (~3% of FLOPs) stays scalar fp32.
// ---------------------------------------------------------------------------

typedef __attribute__((ext_vector_type(16))) __bf16 v16bf;
typedef __attribute__((ext_vector_type(8)))  float  v8f;
typedef __attribute__((ext_vector_type(4)))  unsigned int v4u;
typedef __attribute__((ext_vector_type(8)))  int v8i;
typedef __attribute__((ext_vector_type(4)))  int v4i;

union BFrag { uint4 q[2]; v16bf v; };

__device__ __forceinline__ unsigned short f2bf(float f) {
    unsigned int u = __float_as_uint(f);
    u += 0x7fffu + ((u >> 16) & 1u);      // round-to-nearest-even
    return (unsigned short)(u >> 16);
}

#define ACT_STRIDE   264                  // 256 cols + 8 pad (bf16 elems)
#define ROWS_PER_BLK 128
#define ACT_BYTES    (ROWS_PER_BLK * ACT_STRIDE * 2)
#define WSM_BYTES    (256 * ACT_STRIDE * 2)
#define LDS_BYTES    (ACT_BYTES + WSM_BYTES)

// ---------------------------------------------------------------------------
// TDM: DMA one 256x256 bf16 weight matrix from global into LDS at byte
// offset ACT_BYTES, inserting 16B padding every 512B (-> 264-elem stride).
// D# packing per CDNA5 ISA 8.3/8.4 (2D tensor, groups 2/3 unused).
// ---------------------------------------------------------------------------
__device__ __forceinline__ void tdm_load_w(const unsigned short* gsrc)
{
    unsigned long long ga = (unsigned long long)(uintptr_t)gsrc;
    v4u g0;
    g0.x = 1u;                                    // count=1, user descriptor
    g0.y = (unsigned)ACT_BYTES;                   // lds_addr (dyn LDS base = 0)
    g0.z = (unsigned)(ga & 0xffffffffu);          // global_addr[31:0]
    g0.w = (unsigned)((ga >> 32) & 0x01ffffffu)   // global_addr[56:32]
         | 0x80000000u;                           // type=2 ("image")
    v8i g1;
    g1[0] = (int)((1u << 16)                      // data_size = 2 bytes
                | (1u << 20)                      // pad_enable
                | (6u << 22)                      // pad_interval: 128 dwords (512B)
                | (3u << 25));                    // pad_amount: 4 dwords (16B)
    g1[1] = (int)(256u << 16);                    // tensor_dim0 = 256 (bits 79:48)
    g1[2] = (int)(256u << 16);                    // tensor_dim1 = 256 (bits 111:80)
    g1[3] = (int)(256u << 16);                    // tile_dim0 = 256 (bits 127:112)
    g1[4] = 256;                                  // tile_dim1 = 256 (bits 143:128)
    g1[5] = 256;                                  // tensor_dim0_stride = 256
    g1[6] = 0;
    g1[7] = 0;
    v4i gz4 = {0, 0, 0, 0};
    v8i gz8 = {0, 0, 0, 0, 0, 0, 0, 0};
    __builtin_amdgcn_tensor_load_to_lds(g0, g1, gz4, gz4, gz8, 0);
}

// ---------------------------------------------------------------------------
// Fused relation core: h1 = relu(hq+hs+b1); x3 (GEMM 256x256 + bias + relu);
// final stage reduces over rows into out256[pair][256] via atomicAdd.
// grid.x = 75 pairs * 32 tiles, block = 256 threads (8 waves, 16 rows/wave).
// ---------------------------------------------------------------------------
__global__ __launch_bounds__(256)
void relation_kernel(const float* __restrict__ hq, const float* __restrict__ hs,
                     const float* __restrict__ g1b,
                     const unsigned short* __restrict__ wb,   // 3 x 256x256 bf16
                     const float* __restrict__ b2, const float* __restrict__ b3,
                     const float* __restrict__ b4,
                     float* __restrict__ out256)
{
    extern __shared__ char smem[];
    unsigned short* act = (unsigned short*)smem;                 // 128 x 264 bf16
    unsigned short* wsm = act + ROWS_PER_BLK * ACT_STRIDE;       // 256 x 264 bf16

    const int tid  = threadIdx.x;
    const int pair = blockIdx.x >> 5;     // q*5+n
    const int tile = blockIdx.x & 31;     // 32 tiles of 128 rows = 4096 rows
    const int q    = pair / 5;
    const int n    = pair % 5;

    const int lane   = tid & 31;
    const int wv     = tid >> 5;
    const int h      = lane >> 4;         // lane half
    const int m      = lane & 15;
    const int stripe = wv * 16;           // this wave's 16-row stripe

    // kick off W2 DMA; it overlaps the h1 tile construction below
    if (wv == 0) tdm_load_w(wb);

    // ---- phase 0: build h1 tile (bf16) in LDS ----
    {
        const int row = tid >> 1;                       // wave w fills its own stripe
        const int c0  = (tid & 1) * 128;
        const int pq  = tile * 2 + (row >> 6);
        const int ps  = row & 63;
        const float* hqr = hq + (q * 64 + pq) * 256;
        const float* hsr = hs + (n * 64 + ps) * 256;
        unsigned short* dst = act + row * ACT_STRIDE;
        for (int c = c0; c < c0 + 128; c += 4) {
            float4 a = *(const float4*)(hqr + c);
            float4 b = *(const float4*)(hsr + c);
            float4 g = *(const float4*)(g1b + c);
            float v0 = fmaxf(a.x + b.x + g.x, 0.f);
            float v1 = fmaxf(a.y + b.y + g.y, 0.f);
            float v2 = fmaxf(a.z + b.z + g.z, 0.f);
            float v3 = fmaxf(a.w + b.w + g.w, 0.f);
            *(unsigned int*)(dst + c)     = (unsigned)f2bf(v0) | ((unsigned)f2bf(v1) << 16);
            *(unsigned int*)(dst + c + 2) = (unsigned)f2bf(v2) | ((unsigned)f2bf(v3) << 16);
        }
    }

    for (int s = 0; s < 3; ++s) {
        const float* bb = (s == 0) ? b2 : (s == 1) ? b3 : b4;

        if (wv == 0) __builtin_amdgcn_s_wait_tensorcnt(0);  // W(s) landed in LDS
        __syncthreads();

        // A fragments: 16x32 bf16 per k-tile, ISA 16-bit A layout.
        // lanes 0-15 hold K {0..7,16..23}, lanes 16-31 hold K {8..15,24..31}.
        BFrag a[8];
        const unsigned short* ar = act + (stripe + m) * ACT_STRIDE;
        #pragma unroll
        for (int t = 0; t < 8; ++t) {
            const int c1 = t * 32 + h * 8;
            a[t].q[0] = *(const uint4*)(ar + c1);
            a[t].q[1] = *(const uint4*)(ar + c1 + 16);
        }

        const v8f zf = {0.f, 0.f, 0.f, 0.f, 0.f, 0.f, 0.f, 0.f};
        v8f acc[16];
        #pragma unroll
        for (int j = 0; j < 16; ++j) acc[j] = zf;

        // y[row,o] = sum_c act[row,c] * W[o,c]  -> B[k=c][n=o] = W[o][c]
        #pragma unroll
        for (int j = 0; j < 16; ++j) {
            const unsigned short* br = wsm + (j * 16 + m) * ACT_STRIDE + h * 8;
            #pragma unroll
            for (int t = 0; t < 8; ++t) {
                BFrag b;
                b.q[0] = *(const uint4*)(br + t * 32);
                b.q[1] = *(const uint4*)(br + t * 32 + 16);
                acc[j] = __builtin_amdgcn_wmma_f32_16x16x32_bf16(
                             false, a[t].v, false, b.v, (short)0, acc[j], false, false);
            }
        }

        __syncthreads();                       // all waves done reading wsm
        if (s < 2 && wv == 0)
            tdm_load_w(wb + (s + 1) * 65536);  // next-stage DMA overlaps epilogue

        if (s < 2) {
            // bias + relu, write back in place (A-frags already consumed)
            #pragma unroll
            for (int j = 0; j < 16; ++j) {
                const int col = j * 16 + m;
                const float bj = bb[col];
                #pragma unroll
                for (int r = 0; r < 8; ++r) {
                    const int row = stripe + r + 8 * h;   // D layout: M = r + 8*half
                    act[row * ACT_STRIDE + col] = f2bf(fmaxf(acc[j][r] + bj, 0.f));
                }
            }
        } else {
            // bias + relu + reduce over the 16 rows -> out256[pair][col]
            #pragma unroll
            for (int j = 0; j < 16; ++j) {
                const int col = j * 16 + m;
                const float bj = bb[col];
                float sum = 0.f;
                #pragma unroll
                for (int r = 0; r < 8; ++r) sum += fmaxf(acc[j][r] + bj, 0.f);
                sum += __shfl_xor(sum, 16, 32);           // add rows 8..15 half
                if (h == 0) atomicAdd(&out256[pair * 256 + col], sum);
            }
        }
    }
}

// one-time f32 -> bf16 conversion of g2/g3/g4 weights for the TDM path
__global__ void cvt_w_bf16(const float* __restrict__ W2, const float* __restrict__ W3,
                           const float* __restrict__ W4, unsigned short* __restrict__ wb)
{
    const int i = blockIdx.x * blockDim.x + threadIdx.x;   // 0 .. 3*65536-1
    const float* src = (i < 65536) ? W2 : (i < 131072) ? W3 : W4;
    wb[i] = f2bf(src[i & 65535]);
}

// ---------------------------------------------------------------------------
// CNN stack (scalar fp32; ~3% of total FLOPs)
// ---------------------------------------------------------------------------
__global__ void conv1_pool_bn(const float* __restrict__ sx, const float* __restrict__ qx,
                              const float* __restrict__ w, const float* __restrict__ b,
                              const float* __restrict__ g, const float* __restrict__ bt,
                              const float* __restrict__ mn, const float* __restrict__ vr,
                              float* __restrict__ out)
{
    const int img = blockIdx.x >> 6;
    const int co  = blockIdx.x & 63;
    const float* xin = (img < 5) ? sx + img * 3 * 84 * 84
                                 : qx + (img - 5) * 3 * 84 * 84;
    float wv[3][3][3];
    #pragma unroll
    for (int ci = 0; ci < 3; ++ci)
        #pragma unroll
        for (int ky = 0; ky < 3; ++ky)
            #pragma unroll
            for (int kx = 0; kx < 3; ++kx)
                wv[ci][ky][kx] = w[((co * 3 + ci) * 3 + ky) * 3 + kx];
    const float bias = b[co];
    const float sc = g[co] * rsqrtf(vr[co] + 1e-5f);
    const float sh = bt[co] - mn[co] * sc;
    float* op = out + (img * 64 + co) * 41 * 41;
    for (int px = threadIdx.x; px < 41 * 41; px += blockDim.x) {
        const int oy = px / 41, ox = px % 41;
        float mx = -1e30f;
        #pragma unroll
        for (int dy = 0; dy < 2; ++dy)
            #pragma unroll
            for (int dx = 0; dx < 2; ++dx) {
                const int cy = 2 * oy + dy, cx = 2 * ox + dx;
                float acc = bias;
                for (int ci = 0; ci < 3; ++ci) {
                    const float* ip = xin + ci * 84 * 84 + cy * 84 + cx;
                    #pragma unroll
                    for (int ky = 0; ky < 3; ++ky)
                        #pragma unroll
                        for (int kx = 0; kx < 3; ++kx)
                            acc += ip[ky * 84 + kx] * wv[ci][ky][kx];
                }
                mx = fmaxf(mx, acc);
            }
        op[px] = fmaxf(mx * sc + sh, 0.f);
    }
}

__global__ void conv2_pool_bn(const float* __restrict__ in, const float* __restrict__ w,
                              const float* __restrict__ b,
                              const float* __restrict__ g, const float* __restrict__ bt,
                              const float* __restrict__ mn, const float* __restrict__ vr,
                              float* __restrict__ out)
{
    const int img = blockIdx.x >> 6;
    const int co  = blockIdx.x & 63;
    const float bias = b[co];
    const float sc = g[co] * rsqrtf(vr[co] + 1e-5f);
    const float sh = bt[co] - mn[co] * sc;
    const float* ib = in + img * 64 * 41 * 41;
    const float* wb = w + co * 64 * 9;
    float* op = out + (img * 64 + co) * 19 * 19;
    for (int px = threadIdx.x; px < 19 * 19; px += blockDim.x) {
        const int oy = px / 19, ox = px % 19;
        const int cy = 2 * oy, cx = 2 * ox;
        float a00 = bias, a01 = bias, a10 = bias, a11 = bias;
        for (int ci = 0; ci < 64; ++ci) {
            const float* ip = ib + ci * 41 * 41 + cy * 41 + cx;
            const float* wp = wb + ci * 9;
            float p[4][4];
            #pragma unroll
            for (int yy = 0; yy < 4; ++yy)
                #pragma unroll
                for (int xx = 0; xx < 4; ++xx)
                    p[yy][xx] = ip[yy * 41 + xx];
            #pragma unroll
            for (int ky = 0; ky < 3; ++ky)
                #pragma unroll
                for (int kx = 0; kx < 3; ++kx) {
                    const float wvv = wp[ky * 3 + kx];
                    a00 += p[ky][kx] * wvv;
                    a01 += p[ky][kx + 1] * wvv;
                    a10 += p[ky + 1][kx] * wvv;
                    a11 += p[ky + 1][kx + 1] * wvv;
                }
        }
        const float mx = fmaxf(fmaxf(a00, a01), fmaxf(a10, a11));
        op[px] = fmaxf(mx * sc + sh, 0.f);
    }
}

__global__ void conv_bn(const float* __restrict__ in, const float* __restrict__ w,
                        const float* __restrict__ b,
                        const float* __restrict__ g, const float* __restrict__ bt,
                        const float* __restrict__ mn, const float* __restrict__ vr,
                        float* __restrict__ out, int Hin, int Hout)
{
    const int img = blockIdx.x >> 6;
    const int co  = blockIdx.x & 63;
    const float bias = b[co];
    const float sc = g[co] * rsqrtf(vr[co] + 1e-5f);
    const float sh = bt[co] - mn[co] * sc;
    const float* ib = in + img * 64 * Hin * Hin;
    const float* wb = w + co * 576;
    float* op = out + (img * 64 + co) * Hout * Hout;
    for (int px = threadIdx.x; px < Hout * Hout; px += blockDim.x) {
        const int oy = px / Hout, ox = px % Hout;
        float acc = bias;
        for (int ci = 0; ci < 64; ++ci) {
            const float* ip = ib + ci * Hin * Hin + oy * Hin + ox;
            const float* wp = wb + ci * 9;
            #pragma unroll
            for (int ky = 0; ky < 3; ++ky)
                #pragma unroll
                for (int kx = 0; kx < 3; ++kx)
                    acc += ip[ky * Hin + kx] * wp[ky * 3 + kx];
        }
        op[px] = fmaxf(acc * sc + sh, 0.f);
    }
}

__global__ void avgpool_feat(const float* __restrict__ a4, float* __restrict__ feat)
{
    const int img = blockIdx.x >> 6;
    const int co  = blockIdx.x & 63;
    const float* ip = a4 + (img * 64 + co) * 225;
    float* op = feat + (img * 64 + co) * 64;
    const int p = threadIdx.x;
    if (p < 64) {
        const int i = p >> 3, j = p & 7;
        float s = 0.f;
        for (int dy = -1; dy <= 1; ++dy)
            for (int dx = -1; dx <= 1; ++dx) {
                const int y = 2 * i + dy, x = 2 * j + dx;
                if (y >= 0 && y < 15 && x >= 0 && x < 15) s += ip[y * 15 + x];
            }
        op[p] = s * (1.f / 9.f);
    }
}

// hq[q,p,o] / hs[n,p,o] projection through g1 (66-dim K incl. coord channels)
__global__ __launch_bounds__(256)
void hproj(const float* __restrict__ feat, const float* __restrict__ g1w,
           float* __restrict__ hq, float* __restrict__ hs)
{
    const int img = blockIdx.x >> 6;
    const int p   = blockIdx.x & 63;
    __shared__ float sfeat[64];
    if (threadIdx.x < 64) sfeat[threadIdx.x] = feat[(img * 64 + threadIdx.x) * 64 + p];
    __syncthreads();
    const int o = threadIdx.x;
    const int cofs = (img < 5) ? 0 : 66;
    const float* wr = g1w + o * 132 + cofs;
    float acc = 0.f;
    for (int c = 0; c < 64; ++c) acc += sfeat[c] * wr[c];
    acc += (float)(p >> 3) * 0.125f * wr[64];
    acc += (float)(p & 7)  * 0.125f * wr[65];
    if (img < 5) hs[(img * 64 + p) * 256 + o] = acc;
    else         hq[((img - 5) * 64 + p) * 256 + o] = acc;
}

__global__ void zero_kernel(float* __restrict__ p, int n)
{
    const int i = blockIdx.x * blockDim.x + threadIdx.x;
    if (i < n) p[i] = 0.f;
}

// f1 -> f2 -> f3 -> bn -> f4 -> sigmoid, one block per (q,n) row
__global__ __launch_bounds__(256)
void mlp_kernel(const float* __restrict__ out256,
                const float* __restrict__ f1w, const float* __restrict__ f1b,
                const float* __restrict__ f2w, const float* __restrict__ f2b,
                const float* __restrict__ f3w, const float* __restrict__ f3b,
                const float* __restrict__ fg, const float* __restrict__ fb,
                const float* __restrict__ fm, const float* __restrict__ fv,
                const float* __restrict__ f4w, const float* __restrict__ f4b,
                float* __restrict__ score)
{
    __shared__ float sx[256], s1[256], s2[256], s3[64];
    const int r = blockIdx.x, t = threadIdx.x;
    sx[t] = out256[r * 256 + t];
    __syncthreads();
    {
        const float* wr = f1w + t * 256;
        float a = f1b[t];
        for (int c = 0; c < 256; ++c) a += sx[c] * wr[c];
        s1[t] = fmaxf(a, 0.f);
    }
    __syncthreads();
    {
        const float* wr = f2w + t * 256;
        float a = f2b[t];
        for (int c = 0; c < 256; ++c) a += s1[c] * wr[c];
        s2[t] = fmaxf(a, 0.f);
    }
    __syncthreads();
    if (t < 64) {
        const float* wr = f3w + t * 256;
        float a = f3b[t];
        for (int c = 0; c < 256; ++c) a += s2[c] * wr[c];
        const float sc = fg[t] * rsqrtf(fv[t] + 1e-5f);
        s3[t] = fmaxf(a * sc + fb[t] - fm[t] * sc, 0.f);
    }
    __syncthreads();
    if (t == 0) {
        float z = f4b[0];
        for (int c = 0; c < 64; ++c) z += s3[c] * f4w[c];
        score[r] = 1.f / (1.f + expf(-z));
    }
}

__global__ void final_kernel(const float* __restrict__ score,
                             const int* __restrict__ sy, const int* __restrict__ qy,
                             float* __restrict__ out)
{
    if (threadIdx.x == 0 && blockIdx.x == 0) {
        float loss = 0.f;
        int correct = 0;
        for (int q = 0; q < 15; ++q) {
            float best = -1.f;
            int bi = 0;
            for (int n = 0; n < 5; ++n) {
                const float sc = score[q * 5 + n];
                const float lab = (sy[n] == qy[q]) ? 1.f : 0.f;
                const float d = lab - sc;
                loss += d * d;
                if (sc > best) { best = sc; bi = n; }   // first-max, argmax semantics
            }
            const int pred = sy[bi];
            out[1 + q] = (float)pred;
            if (pred == qy[q]) ++correct;
        }
        out[0]  = loss;
        out[16] = (float)correct;
    }
}

// ---------------------------------------------------------------------------
extern "C" void kernel_launch(void* const* d_in, const int* in_sizes, int n_in,
                              void* d_out, int out_size, void* d_ws, size_t ws_size,
                              hipStream_t stream)
{
    (void)in_sizes; (void)n_in; (void)out_size; (void)ws_size;

    const float* sx = (const float*)d_in[0];
    const int*   sy = (const int*)  d_in[1];
    const float* qx = (const float*)d_in[2];
    const int*   qy = (const int*)  d_in[3];
    const float* w1 = (const float*)d_in[4],  *b1 = (const float*)d_in[5];
    const float* bn1g = (const float*)d_in[6],  *bn1b = (const float*)d_in[7];
    const float* bn1m = (const float*)d_in[8],  *bn1v = (const float*)d_in[9];
    const float* w2 = (const float*)d_in[10], *b2 = (const float*)d_in[11];
    const float* bn2g = (const float*)d_in[12], *bn2b = (const float*)d_in[13];
    const float* bn2m = (const float*)d_in[14], *bn2v = (const float*)d_in[15];
    const float* w3 = (const float*)d_in[16], *b3 = (const float*)d_in[17];
    const float* bn3g = (const float*)d_in[18], *bn3b = (const float*)d_in[19];
    const float* bn3m = (const float*)d_in[20], *bn3v = (const float*)d_in[21];
    const float* w4 = (const float*)d_in[22], *b4 = (const float*)d_in[23];
    const float* bn4g = (const float*)d_in[24], *bn4b = (const float*)d_in[25];
    const float* bn4m = (const float*)d_in[26], *bn4v = (const float*)d_in[27];
    const float* g1w = (const float*)d_in[28], *g1b = (const float*)d_in[29];
    const float* g2w = (const float*)d_in[30], *g2b = (const float*)d_in[31];
    const float* g3w = (const float*)d_in[32], *g3b = (const float*)d_in[33];
    const float* g4w = (const float*)d_in[34], *g4b = (const float*)d_in[35];
    const float* f1w = (const float*)d_in[36], *f1b = (const float*)d_in[37];
    const float* f2w = (const float*)d_in[38], *f2b = (const float*)d_in[39];
    const float* f3w = (const float*)d_in[40], *f3b = (const float*)d_in[41];
    const float* f4w = (const float*)d_in[42], *f4b = (const float*)d_in[43];
    const float* fbg = (const float*)d_in[44], *fbb = (const float*)d_in[45];
    const float* fbm = (const float*)d_in[46], *fbv = (const float*)d_in[47];

    // workspace layout (floats); total ~15.2 MB
    float* ws     = (float*)d_ws;
    float* A1     = ws;                       // 20*64*41*41
    float* A2     = A1 + 20 * 64 * 41 * 41;   // 20*64*19*19
    float* A3     = A2 + 20 * 64 * 19 * 19;   // 20*64*17*17
    float* A4     = A3 + 20 * 64 * 17 * 17;   // 20*64*15*15
    float* FEAT   = A4 + 20 * 64 * 15 * 15;   // 20*64*64
    float* HQ     = FEAT + 20 * 64 * 64;      // 15*64*256
    float* HS     = HQ + 15 * 64 * 256;       // 5*64*256
    float* OUT256 = HS + 5 * 64 * 256;        // 75*256
    float* SCORE  = OUT256 + 75 * 256;        // 75
    unsigned short* WB = (unsigned short*)(SCORE + 75 + 1);   // 3*65536 bf16

    // CNN feature stack (20 images)
    conv1_pool_bn<<<20 * 64, 256, 0, stream>>>(sx, qx, w1, b1, bn1g, bn1b, bn1m, bn1v, A1);
    conv2_pool_bn<<<20 * 64, 256, 0, stream>>>(A1, w2, b2, bn2g, bn2b, bn2m, bn2v, A2);
    conv_bn<<<20 * 64, 256, 0, stream>>>(A2, w3, b3, bn3g, bn3b, bn3m, bn3v, A3, 19, 17);
    conv_bn<<<20 * 64, 256, 0, stream>>>(A3, w4, b4, bn4g, bn4b, bn4m, bn4v, A4, 17, 15);
    avgpool_feat<<<20 * 64, 64, 0, stream>>>(A4, FEAT);

    // g1 projection to hq/hs
    hproj<<<20 * 64, 256, 0, stream>>>(FEAT, g1w, HQ, HS);

    // bf16 weights for TDM staging; zero the relation accumulators
    cvt_w_bf16<<<(3 * 65536) / 256, 256, 0, stream>>>(g2w, g3w, g4w, WB);
    zero_kernel<<<(75 * 256 + 255) / 256, 256, 0, stream>>>(OUT256, 75 * 256);

    // fused WMMA+TDM relation core (needs >64KB dynamic LDS -> raise the cap)
    (void)hipFuncSetAttribute(reinterpret_cast<const void*>(&relation_kernel),
                              hipFuncAttributeMaxDynamicSharedMemorySize, (int)LDS_BYTES);
    relation_kernel<<<75 * 32, 256, LDS_BYTES, stream>>>(
        HQ, HS, g1b, WB, b2, b3, b4, OUT256);

    // MLP head + sigmoid
    mlp_kernel<<<75, 256, 0, stream>>>(OUT256, f1w, f1b, f2w, f2b, f3w, f3b,
                                       fbg, fbb, fbm, fbv, f4w, f4b, SCORE);

    // loss / pred / correct -> d_out (17 elems: loss, 15 preds, correct)
    final_kernel<<<1, 32, 0, stream>>>(SCORE, sy, qy, (float*)d_out);
}
